// RGA_direct_with_charge_encode_15401752723468
// MI455X (gfx1250) — matrix operand order repaired and checked by reading, not compiled, not run on hardware
//
#include <hip/hip_runtime.h>

// ============================================================================
// RGA GATv2 + MLP forward for MI455X (gfx1250, wave32, WMMA).
// GEMMs: bf16 A/W, async-DMA (global_load_async_to_lds_b128) double-buffered
// LDS tiles, v_wmma_f32_16x16x32_bf16 accumulation.
// Edge segment-softmax/aggregation: VALU + global float atomics (L2-resident).
// ============================================================================

typedef __bf16 bf16;
typedef bf16  v16bf __attribute__((ext_vector_type(16)));
typedef float v8f   __attribute__((ext_vector_type(8)));

#define LDA 40   // LDS row pitch in bf16: 80B = 16B-aligned rows, bank-stride 20

static inline int nblk(long long n) { return (int)((n + 255) / 256); }

__device__ __forceinline__ void atomicMaxF(float* addr, float val) {
  if (val >= 0.0f) atomicMax((int*)addr, __float_as_int(val));
  else             atomicMin((unsigned int*)addr, __float_as_uint(val));
}

// Async DMA: copy 16B global -> LDS, tracked by ASYNCcnt.
__device__ __forceinline__ void async_cp16(void* lds, const void* gmem) {
  unsigned l = (unsigned)(unsigned long long)(uintptr_t)lds;          // LDS byte offset
  unsigned long long g = (unsigned long long)(uintptr_t)gmem;
  asm volatile("global_load_async_to_lds_b128 %0, %1, off" :: "v"(l), "v"(g) : "memory");
}

// ---------------------------------------------------------------------------
// Elementwise / BN kernels
// ---------------------------------------------------------------------------
__global__ void k_fill(float* __restrict__ p, float v, long long n) {
  long long i = (long long)blockIdx.x * blockDim.x + threadIdx.x;
  if (i < n) p[i] = v;
}

__global__ void k_bn_stats(const float* __restrict__ x, int rows, int cols,
                           float* __restrict__ acc) {
  int col = blockIdx.x * blockDim.x + threadIdx.x;
  if (col >= cols) return;
  int r0 = blockIdx.y * 256;
  int r1 = r0 + 256; if (r1 > rows) r1 = rows;
  float s = 0.f, s2 = 0.f;
  for (int r = r0; r < r1; ++r) {
    float v = x[(size_t)r * cols + col];
    s += v; s2 += v * v;
  }
  atomicAdd(&acc[col], s);
  atomicAdd(&acc[cols + col], s2);
}

__global__ void k_bn_finalize(const float* __restrict__ acc, const float* __restrict__ gamma,
                              const float* __restrict__ beta, int rows, int cols,
                              float* __restrict__ scale, float* __restrict__ shift) {
  int c = blockIdx.x * blockDim.x + threadIdx.x;
  if (c >= cols) return;
  float inv = 1.0f / (float)rows;
  float m   = acc[c] * inv;
  float var = acc[cols + c] * inv - m * m;
  float sc  = gamma[c] * rsqrtf(var + 1e-5f);
  scale[c] = sc;
  shift[c] = beta[c] - m * sc;
}

// v = g*scale+shift (+res), optional relu; writes f32 (if out) and bf16 (if out16)
__global__ void k_bn_apply(const float* __restrict__ g, const float* __restrict__ scale,
                           const float* __restrict__ shift, const float* __restrict__ res,
                           float* __restrict__ out, bf16* __restrict__ out16,
                           int rows, int cols, int ostride, int relu) {
  long long idx = (long long)blockIdx.x * blockDim.x + threadIdx.x;
  if (idx >= (long long)rows * cols) return;
  int r = (int)(idx / cols);
  int c = (int)(idx - (long long)r * cols);
  float v = g[idx] * scale[c] + shift[c];
  if (res) v += res[idx];
  if (relu) v = fmaxf(v, 0.0f);
  if (out)   out[(size_t)r * ostride + c] = v;
  if (out16) out16[(size_t)r * ostride + c] = (bf16)v;
}

// W[K,Nc] f32 -> Wt[Nc,K] bf16 (one-time per GEMM; makes tiles DMA-able)
__global__ void k_wcvt(const float* __restrict__ W, bf16* __restrict__ Wt, int K, int Nc) {
  long long idx = (long long)blockIdx.x * blockDim.x + threadIdx.x;
  if (idx >= (long long)K * Nc) return;
  int k = (int)(idx / Nc);
  int c = (int)(idx - (long long)k * Nc);
  Wt[(size_t)c * K + k] = (bf16)W[idx];
}

// ---------------------------------------------------------------------------
// WMMA GEMM: C[M,Nc] = A[M,K](bf16) * W[K,Nc] (given as Wt[Nc,K] bf16) (+bias)
// Block: 256 thr = 8 waves; tile 64x64; wave tile 16x32 (two 16x16x32 WMMAs).
// Tiles staged via global_load_async_to_lds_b128 (16B/lane), double-buffered:
// issue tile t+1, s_wait_asynccnt 2 (tile t landed), barrier, WMMA tile t.
// OOB rows/cols clamped on load (branch-free), guarded on store.
// ---------------------------------------------------------------------------
template <bool OUT_BF16>
__global__ __launch_bounds__(256)
void k_gemm_wmma(const bf16* __restrict__ A, const bf16* __restrict__ Wt,
                 const float* __restrict__ bias, void* __restrict__ outp,
                 int M, int K, int Nc) {
  __shared__ __align__(16) bf16 As[2][64][LDA];
  __shared__ __align__(16) bf16 Bs[2][64][LDA];   // Bs[buf][out-col][k]

  const int tid   = threadIdx.x;
  const int lane  = tid & 31;
  const int wave  = tid >> 5;
  const int half  = lane >> 4;   // 0: lanes 0-15, 1: lanes 16-31
  const int lr    = lane & 15;
  const int waveM = wave & 3;    // 4 row-waves
  const int waveN = wave >> 2;   // 2 col-waves
  const int rowBase = blockIdx.y * 64;
  const int colBase = blockIdx.x * 64;

  // Per-thread 16B copy assignments (clamped in-bounds; OOB rows/cols only feed
  // output elements that are never stored).
  const int ar = tid >> 2, ac = (tid & 3) * 8;      // A tile: row 0..63, col 0/8/16/24
  int agr = rowBase + ar; if (agr >= M) agr = M - 1;
  const bf16* aSrc = A + (size_t)agr * K + ac;
  const int bc = tid >> 2, bk = (tid & 3) * 8;      // B tile: out-col 0..63, k 0/8/16/24
  int bgc = colBase + bc; if (bgc >= Nc) bgc = Nc - 1;
  const bf16* bSrc = Wt + (size_t)bgc * K + bk;

  v8f acc0 = {0.f, 0.f, 0.f, 0.f, 0.f, 0.f, 0.f, 0.f};
  v8f acc1 = {0.f, 0.f, 0.f, 0.f, 0.f, 0.f, 0.f, 0.f};

  const int T = K >> 5;                  // K is a multiple of 32 (64 or 256)
  async_cp16(&As[0][ar][ac], aSrc);      // prologue: tile 0 -> buffer 0
  async_cp16(&Bs[0][bc][bk], bSrc);

  for (int t = 0; t < T; ++t) {
    const int cur = t & 1;
    if (t + 1 < T) {
      const int k0n = (t + 1) << 5;
      async_cp16(&As[cur ^ 1][ar][ac], aSrc + k0n);
      async_cp16(&Bs[cur ^ 1][bc][bk], bSrc + k0n);
      asm volatile("s_wait_asynccnt 0x2" ::: "memory");   // tile t done, t+1 in flight
    } else {
      asm volatile("s_wait_asynccnt 0x0" ::: "memory");
    }
    __syncthreads();

    // A fragment (16x32): lanes 0-15 K=0..15, lanes 16-31 K=16..31; VGPR i -> K=2i,2i+1
    v16bf a, b0, b1;
#pragma unroll
    for (int i = 0; i < 8; ++i) {
      int kb = 2 * i + 16 * half;
      a[2 * i]     = As[cur][waveM * 16 + lr][kb];
      a[2 * i + 1] = As[cur][waveM * 16 + lr][kb + 1];
    }
    // B fragment (32x16): lanes 0-15 K=0..15, lanes 16-31 K=16..31; VGPR v -> K=2v,2v+1
#pragma unroll
    for (int v = 0; v < 8; ++v) {
      int kb = 2 * v + 16 * half;
      b0[2 * v]     = Bs[cur][waveN * 32 + lr][kb];
      b0[2 * v + 1] = Bs[cur][waveN * 32 + lr][kb + 1];
      b1[2 * v]     = Bs[cur][waveN * 32 + 16 + lr][kb];
      b1[2 * v + 1] = Bs[cur][waveN * 32 + 16 + lr][kb + 1];
    }
    acc0 = __builtin_amdgcn_wmma_f32_16x16x32_bf16(false, a, false, b0, (short)0, acc0, false, false);
    acc1 = __builtin_amdgcn_wmma_f32_16x16x32_bf16(false, a, false, b1, (short)0, acc1, false, false);
    __syncthreads();   // everyone done reading buf[cur] before it is re-filled
  }

  // C/D layout: VGPR r -> (M=r, N=lane) lanes 0-15; (M=8+r, N=lane-16) lanes 16-31
#pragma unroll
  for (int r = 0; r < 8; ++r) {
    int row = rowBase + waveM * 16 + half * 8 + r;
    if (row >= M) continue;
    int col0 = colBase + waveN * 32 + lr;
    int col1 = col0 + 16;
    if (col0 < Nc) {
      float v = acc0[r]; if (bias) v += bias[col0];
      if (OUT_BF16) ((bf16*)outp)[(size_t)row * Nc + col0] = (bf16)v;
      else          ((float*)outp)[(size_t)row * Nc + col0] = v;
    }
    if (col1 < Nc) {
      float v = acc1[r]; if (bias) v += bias[col1];
      if (OUT_BF16) ((bf16*)outp)[(size_t)row * Nc + col1] = (bf16)v;
      else          ((float*)outp)[(size_t)row * Nc + col1] = v;
    }
  }
}

// ---------------------------------------------------------------------------
// gatc transforms (K=1 outer product): out[n,j] = c0[n]*w[j] + b[j], HC=128
// ---------------------------------------------------------------------------
__global__ void k_outer128(const float* __restrict__ c0, const float* __restrict__ w,
                           const float* __restrict__ b, bf16* __restrict__ out, int Nn) {
  long long idx = (long long)blockIdx.x * blockDim.x + threadIdx.x;
  if (idx >= (long long)Nn * 128) return;
  int n = (int)(idx >> 7), j = (int)(idx & 127);
  out[idx] = (bf16)(c0[n] * w[j] + b[j]);
}

// ---------------------------------------------------------------------------
// Edge kernels (H=4). One wave per (edge, head) for the logit; edge_attr @ We
// fused in (K=8). Segment max/sum via float atomics.
// ---------------------------------------------------------------------------
__global__ __launch_bounds__(256)
void k_edge_logits(const bf16* __restrict__ XL, const bf16* __restrict__ XR,
                   const float* __restrict__ ea, const float* __restrict__ We,
                   const float* __restrict__ att, const int* __restrict__ src,
                   const int* __restrict__ dst, int E, int C,
                   float* __restrict__ logits, float* __restrict__ maxb) {
  long long gw = ((long long)blockIdx.x * blockDim.x + threadIdx.x) >> 5;
  int lane = threadIdx.x & 31;
  if (gw >= (long long)E * 4) return;
  int e = (int)(gw >> 2), h = (int)(gw & 3);
  int sN = src[e], dN = dst[e];
  int HC = 4 * C;
  float eav[8];
#pragma unroll
  for (int k = 0; k < 8; ++k) eav[k] = ea[(size_t)e * 8 + k];
  const bf16*  xlp  = XL + (size_t)sN * HC + h * C;
  const bf16*  xrp  = XR + (size_t)dN * HC + h * C;
  const float* attp = att + h * C;
  const float* Wep  = We + h * C;
  float acc = 0.f;
  for (int c = lane; c < C; c += 32) {
    float ec = 0.f;
#pragma unroll
    for (int k = 0; k < 8; ++k) ec += eav[k] * Wep[(size_t)k * HC + c];
    float m = (float)xlp[c] + (float)xrp[c] + ec;
    m = (m > 0.f) ? m : 0.2f * m;          // leaky_relu(0.2)
    acc += attp[c] * m;
  }
#pragma unroll
  for (int off = 16; off > 0; off >>= 1) acc += __shfl_down(acc, off, 32);
  if (lane == 0) {
    logits[(size_t)e * 4 + h] = acc;
    atomicMaxF(&maxb[(size_t)dN * 4 + h], acc);
  }
}

__global__ void k_edge_exp(const float* __restrict__ logits, const int* __restrict__ dst,
                           const float* __restrict__ maxb, float* __restrict__ exb,
                           float* __restrict__ den, int E) {
  long long idx = (long long)blockIdx.x * blockDim.x + threadIdx.x;
  if (idx >= (long long)E * 4) return;
  int e = (int)(idx >> 2), h = (int)(idx & 3);
  int dN = dst[e];
  float v = __expf(logits[idx] - maxb[(size_t)dN * 4 + h]);
  exb[idx] = v;
  atomicAdd(&den[(size_t)dN * 4 + h], v);
}

// exb <- alpha/H  (fold the 1/H head-mean in here)
__global__ void k_alpha(float* __restrict__ exb, const int* __restrict__ dst,
                        const float* __restrict__ den, int E) {
  long long idx = (long long)blockIdx.x * blockDim.x + threadIdx.x;
  if (idx >= (long long)E * 4) return;
  int e = (int)(idx >> 2), h = (int)(idx & 3);
  exb[idx] = 0.25f * exb[idx] / den[(size_t)dst[e] * 4 + h];
}

// agg[dst,c] += sum_h alpha[e,h]/H * XL[src, h*C + c]
__global__ void k_edge_agg(const bf16* __restrict__ XL, const float* __restrict__ exb,
                           const int* __restrict__ src, const int* __restrict__ dst,
                           int E, int C, int cshift, float* __restrict__ agg) {
  long long idx = (long long)blockIdx.x * blockDim.x + threadIdx.x;
  if (idx >= (long long)E * C) return;
  int e = (int)(idx >> cshift);
  int c = (int)(idx & (C - 1));
  int sN = src[e], dN = dst[e];
  const bf16* xlp = XL + (size_t)sN * 4 * C;
  const float* al = exb + (size_t)e * 4;
  float v = al[0] * (float)xlp[c] + al[1] * (float)xlp[C + c] +
            al[2] * (float)xlp[2 * C + c] + al[3] * (float)xlp[3 * C + c];
  atomicAdd(&agg[(size_t)dN * C + c], v);
}

// ---------------------------------------------------------------------------
// Head kernels
// ---------------------------------------------------------------------------
__global__ void k_pair(const float* __restrict__ latent, const int* __restrict__ t0,
                       const int* __restrict__ t1, bf16* __restrict__ pair, int P) {
  long long idx = (long long)blockIdx.x * blockDim.x + threadIdx.x;
  if (idx >= (long long)P * 64) return;
  int p = (int)(idx >> 6), c = (int)(idx & 63);
  pair[idx] = (bf16)(latent[(size_t)t0[p] * 64 + c] + latent[(size_t)t1[p] * 64 + c]);
}

__global__ void k_fc6(const float* __restrict__ y, const float* __restrict__ w,
                      const float* __restrict__ b, float* __restrict__ out, int P) {
  int p = blockIdx.x * blockDim.x + threadIdx.x;
  if (p >= P) return;
  float a = b[0];
  const float* yr = y + (size_t)p * 32;
#pragma unroll
  for (int k = 0; k < 32; ++k) a += yr[k] * w[k];
  out[p] = fmaxf(a, 0.0f);
}

// ---------------------------------------------------------------------------
// Host-side orchestration
// ---------------------------------------------------------------------------
struct GatP { const float *Wl, *bl, *Wr, *br, *We, *att, *bias; };
struct BnP  { const float *g, *b; };

template <bool OB>
static void run_gemm(const bf16* A, const float* W, const float* bias, void* out,
                     int M, int K, int Nc, bf16* wtbuf, hipStream_t s) {
  k_wcvt<<<nblk((long long)K * Nc), 256, 0, s>>>(W, wtbuf, K, Nc);
  dim3 g((Nc + 63) / 64, (M + 63) / 64);
  k_gemm_wmma<OB><<<g, 256, 0, s>>>(A, wtbuf, bias, out, M, K, Nc);
}

static void run_bn(const float* g, int rows, int cols, const float* gamma, const float* beta,
                   float* bnacc, float* scale, float* shift, hipStream_t s) {
  k_fill<<<nblk(2 * cols), 256, 0, s>>>(bnacc, 0.f, 2 * cols);
  dim3 gs((cols + 255) / 256, (rows + 255) / 256);
  k_bn_stats<<<gs, 256, 0, s>>>(g, rows, cols, bnacc);
  k_bn_finalize<<<nblk(cols), 256, 0, s>>>(bnacc, gamma, beta, rows, cols, scale, shift);
}

static void run_gat_edges(const bf16* XL, const bf16* XR, const float* ea, const GatP& gp,
                          const int* src, const int* dst, int E, int Nn, int C,
                          float* logits, float* exb, float* maxb, float* den, float* agg,
                          hipStream_t s) {
  k_fill<<<nblk((long long)Nn * 4), 256, 0, s>>>(maxb, -__builtin_inff(), (long long)Nn * 4);
  k_fill<<<nblk((long long)Nn * 4), 256, 0, s>>>(den, 0.f, (long long)Nn * 4);
  k_fill<<<nblk((long long)Nn * C), 256, 0, s>>>(agg, 0.f, (long long)Nn * C);
  k_edge_logits<<<nblk((long long)E * 4 * 32), 256, 0, s>>>(XL, XR, ea, gp.We, gp.att, src, dst,
                                                            E, C, logits, maxb);
  k_edge_exp<<<nblk((long long)E * 4), 256, 0, s>>>(logits, dst, maxb, exb, den, E);
  k_alpha<<<nblk((long long)E * 4), 256, 0, s>>>(exb, dst, den, E);
  int cshift = (C == 256) ? 8 : 5;
  k_edge_agg<<<nblk((long long)E * C), 256, 0, s>>>(XL, exb, src, dst, E, C, cshift, agg);
}

extern "C" void kernel_launch(void* const* d_in, const int* in_sizes, int n_in,
                              void* d_out, int out_size, void* d_ws, size_t ws_size,
                              hipStream_t stream) {
  (void)n_in; (void)out_size; (void)ws_size;
  const float* x         = (const float*)d_in[0];
  const float* charge    = (const float*)d_in[1];
  const float* edge_attr = (const float*)d_in[2];
  const int*   edge_idx  = (const int*)d_in[3];
  const int*   targ_idx  = (const int*)d_in[4];
  const int N = in_sizes[0] / 64;
  const int E = in_sizes[2] / 8;
  const int P = in_sizes[4] / 2;
  const int* esrc = edge_idx;
  const int* edst = edge_idx + E;
  const int* t0   = targ_idx;
  const int* t1   = targ_idx + P;

  // ---- params (setup_inputs dict insertion order, nested dicts depth-first) ----
  int pi = 5;
  auto nextp  = [&]() { return (const float*)d_in[pi++]; };
  auto nextbn = [&]() { BnP b; b.g = nextp(); b.b = nextp(); return b; };
  auto nextgat = [&]() {
    GatP g;
    g.Wl = nextp(); g.bl = nextp(); g.Wr = nextp(); g.br = nextp();
    g.We = nextp(); g.att = nextp(); g.bias = nextp();  // bias cancels in following BN
    return g;
  };
  BnP top_bn = nextbn();
  GatP gat1 = nextgat(); BnP bn1 = nextbn();
  GatP gat2 = nextgat(); BnP bn2 = nextbn();
  GatP gat3 = nextgat(); BnP bn3 = nextbn();
  GatP gat4 = nextgat(); BnP bn4 = nextbn();
  GatP gat5 = nextgat(); BnP bn5 = nextbn();
  BnP charge_bn = nextbn();
  GatP gatc = nextgat(); BnP bnc = nextbn();
  const float *fcw[6], *fcb[6]; BnP hbn[5];
  for (int i = 0; i < 5; ++i) { fcw[i] = nextp(); fcb[i] = nextp(); hbn[i] = nextbn(); }
  fcw[5] = nextp(); fcb[5] = nextp();

  // ---- workspace carve ----
  char* wp = (char*)d_ws;
  auto alloc = [&](size_t bytes) { void* p = wp; wp += (bytes + 255) & ~(size_t)255; return p; };
  bf16*  h016   = (bf16*)alloc((size_t)N * 64 * 2);
  float* hA     = (float*)alloc((size_t)N * 256 * 4);
  float* hB     = (float*)alloc((size_t)N * 256 * 4);
  bf16*  hA16   = (bf16*)alloc((size_t)N * 256 * 2);
  bf16*  hB16   = (bf16*)alloc((size_t)N * 256 * 2);
  bf16*  XL     = (bf16*)alloc((size_t)N * 1024 * 2);
  bf16*  XR     = (bf16*)alloc((size_t)N * 1024 * 2);
  float* agg    = (float*)alloc((size_t)N * 256 * 4);
  float* logits = (float*)alloc((size_t)E * 4 * 4);
  float* exb    = (float*)alloc((size_t)E * 4 * 4);
  float* maxb   = (float*)alloc((size_t)N * 4 * 4);
  float* den    = (float*)alloc((size_t)N * 4 * 4);
  float* c0     = (float*)alloc((size_t)N * 4);
  bf16*  pair16 = (bf16*)alloc((size_t)P * 64 * 2);
  float* yA     = (float*)alloc((size_t)P * 256 * 4);
  bf16*  yA16   = (bf16*)alloc((size_t)P * 256 * 2);
  float* yB     = (float*)alloc((size_t)P * 256 * 4);
  float* y5     = (float*)alloc((size_t)P * 32 * 4);
  bf16*  wtbuf  = (bf16*)alloc((size_t)256 * 1024 * 2);   // max K*Nc transposed weight
  float* bnacc  = (float*)alloc(2048 * 4);
  float* scaleb = (float*)alloc(1024 * 4);
  float* shiftb = (float*)alloc(1024 * 4);

  float* latent = (float*)d_out;              // [N,64]
  float* dis    = latent + (size_t)N * 64;    // [P]

  // ---- top BN -> h0 (bf16 only; consumed solely by gat1 GEMMs) ----
  run_bn(x, N, 64, top_bn.g, top_bn.b, bnacc, scaleb, shiftb, stream);
  k_bn_apply<<<nblk((long long)N * 64), 256, 0, stream>>>(x, scaleb, shiftb, nullptr,
                                                          nullptr, h016, N, 64, 64, 0);

  // ---- gat1: 64 -> 256 ----
  run_gemm<true>(h016, gat1.Wl, gat1.bl, XL, N, 64, 1024, wtbuf, stream);
  run_gemm<true>(h016, gat1.Wr, gat1.br, XR, N, 64, 1024, wtbuf, stream);
  run_gat_edges(XL, XR, edge_attr, gat1, esrc, edst, E, N, 256, logits, exb, maxb, den, agg, stream);
  run_bn(agg, N, 256, bn1.g, bn1.b, bnacc, scaleb, shiftb, stream);
  k_bn_apply<<<nblk((long long)N * 256), 256, 0, stream>>>(agg, scaleb, shiftb, nullptr,
                                                           hA, hA16, N, 256, 256, 1);

  // ---- residual gat2..gat4: 256 -> 256 ----
  const GatP* gg[3] = { &gat2, &gat3, &gat4 };
  const BnP*  bb[3] = { &bn2, &bn3, &bn4 };
  for (int i = 0; i < 3; ++i) {
    run_gemm<true>(hA16, gg[i]->Wl, gg[i]->bl, XL, N, 256, 1024, wtbuf, stream);
    run_gemm<true>(hA16, gg[i]->Wr, gg[i]->br, XR, N, 256, 1024, wtbuf, stream);
    run_gat_edges(XL, XR, edge_attr, *gg[i], esrc, edst, E, N, 256, logits, exb, maxb, den, agg, stream);
    run_bn(agg, N, 256, bb[i]->g, bb[i]->b, bnacc, scaleb, shiftb, stream);
    k_bn_apply<<<nblk((long long)N * 256), 256, 0, stream>>>(agg, scaleb, shiftb, hA,
                                                             hB, hB16, N, 256, 256, 1);
    float* tf = hA; hA = hB; hB = tf;
    bf16*  tb = hA16; hA16 = hB16; hB16 = tb;
  }

  // ---- gat5: 256 -> 32, into latent[:, 0:32] ----
  run_gemm<true>(hA16, gat5.Wl, gat5.bl, XL, N, 256, 128, wtbuf, stream);
  run_gemm<true>(hA16, gat5.Wr, gat5.br, XR, N, 256, 128, wtbuf, stream);
  run_gat_edges(XL, XR, edge_attr, gat5, esrc, edst, E, N, 32, logits, exb, maxb, den, agg, stream);
  run_bn(agg, N, 32, bn5.g, bn5.b, bnacc, scaleb, shiftb, stream);
  k_bn_apply<<<nblk((long long)N * 32), 256, 0, stream>>>(agg, scaleb, shiftb, nullptr,
                                                          latent, nullptr, N, 32, 64, 1);

  // ---- charge branch: BN -> gatc (K=1 outer product) -> latent[:, 32:64] ----
  run_bn(charge, N, 1, charge_bn.g, charge_bn.b, bnacc, scaleb, shiftb, stream);
  k_bn_apply<<<nblk(N), 256, 0, stream>>>(charge, scaleb, shiftb, nullptr, c0, nullptr, N, 1, 1, 0);
  k_outer128<<<nblk((long long)N * 128), 256, 0, stream>>>(c0, gatc.Wl, gatc.bl, XL, N);
  k_outer128<<<nblk((long long)N * 128), 256, 0, stream>>>(c0, gatc.Wr, gatc.br, XR, N);
  run_gat_edges(XL, XR, edge_attr, gatc, esrc, edst, E, N, 32, logits, exb, maxb, den, agg, stream);
  run_bn(agg, N, 32, bnc.g, bnc.b, bnacc, scaleb, shiftb, stream);
  k_bn_apply<<<nblk((long long)N * 32), 256, 0, stream>>>(agg, scaleb, shiftb, nullptr,
                                                          latent + 32, nullptr, N, 32, 64, 1);

  // ---- pair head ----
  k_pair<<<nblk((long long)P * 64), 256, 0, stream>>>(latent, t0, t1, pair16, P);

  run_gemm<false>(pair16, fcw[0], fcb[0], yB, P, 64, 256, wtbuf, stream);
  run_bn(yB, P, 256, hbn[0].g, hbn[0].b, bnacc, scaleb, shiftb, stream);
  k_bn_apply<<<nblk((long long)P * 256), 256, 0, stream>>>(yB, scaleb, shiftb, nullptr,
                                                           yA, yA16, P, 256, 256, 1);

  for (int i = 1; i < 4; ++i) {   // fc2..fc4 residual
    run_gemm<false>(yA16, fcw[i], fcb[i], yB, P, 256, 256, wtbuf, stream);
    run_bn(yB, P, 256, hbn[i].g, hbn[i].b, bnacc, scaleb, shiftb, stream);
    k_bn_apply<<<nblk((long long)P * 256), 256, 0, stream>>>(yB, scaleb, shiftb, yA,
                                                             yA, yA16, P, 256, 256, 1);
  }

  run_gemm<false>(yA16, fcw[4], fcb[4], yB, P, 256, 32, wtbuf, stream);  // fc5
  run_bn(yB, P, 32, hbn[4].g, hbn[4].b, bnacc, scaleb, shiftb, stream);
  k_bn_apply<<<nblk((long long)P * 32), 256, 0, stream>>>(yB, scaleb, shiftb, nullptr,
                                                          y5, nullptr, P, 32, 32, 1);

  k_fc6<<<nblk(P), 256, 0, stream>>>(y5, fcw[5], fcb[5], dis, P);
}
